// Swin_Final_61529701482886
// MI455X (gfx1250) — compile-verified, matrix operand order checked by reading, not run on hardware
//
#include <hip/hip_runtime.h>
#include <hip/hip_bf16.h>
#include <math.h>

typedef __attribute__((ext_vector_type(16))) __bf16 v16bf;
typedef __attribute__((ext_vector_type(8)))  __bf16 v8bf;
typedef __attribute__((ext_vector_type(8)))  float  v8f;

// ---------------------------------------------------------------- utilities
__device__ __forceinline__ float gelu_f(float x) {
    return 0.5f * x * (1.0f + erff(x * 0.70710678118654752f));
}
__device__ __forceinline__ int region_of(int z, int dp, int ws, int ss) {
    return (z < dp - ws) ? 0 : ((z < dp - ss) ? 1 : 2);
}

// ---------------------------------------------------------------- patch embed
// Conv3d(1, 24, k=2, s=2): out (b, D/2,D/2,D/2, 24) token-major fp32
__global__ void patch_embed_kernel(const float* __restrict__ x,
                                   const float* __restrict__ w,
                                   const float* __restrict__ bias,
                                   float* __restrict__ out,
                                   int B, int Din, int C) {
    int Dh = Din >> 1;
    long total = (long)B * Dh * Dh * Dh * C;
    long idx = (long)blockIdx.x * blockDim.x + threadIdx.x;
    if (idx >= total) return;
    int c = (int)(idx % C);
    long t = idx / C;
    int xq = (int)(t % Dh);
    int yq = (int)((t / Dh) % Dh);
    int zq = (int)((t / ((long)Dh * Dh)) % Dh);
    int bq = (int)(t / ((long)Dh * Dh * Dh));
    float s = bias[c];
#pragma unroll
    for (int i = 0; i < 2; ++i)
#pragma unroll
        for (int j = 0; j < 2; ++j)
#pragma unroll
            for (int k = 0; k < 2; ++k) {
                float wv = w[((c * 2 + i) * 2 + j) * 2 + k];
                long src = (((long)bq * Din + (2 * zq + i)) * Din + (2 * yq + j)) * Din + (2 * xq + k);
                s += wv * x[src];
            }
    out[t * C + c] = s;
}

// ---------------------------------------------------------------- weight convert
// fp32 (Nr x Kr) row-major -> bf16 (Np x Kp) zero-padded
__global__ void conv_weight_kernel(const float* __restrict__ src, __bf16* __restrict__ dst,
                                   int Nr, int Kr, int Np, int Kp) {
    int i = blockIdx.x * blockDim.x + threadIdx.x;
    if (i >= Np * Kp) return;
    int n = i / Kp, k = i - n * Kp;
    float v = (n < Nr && k < Kr) ? src[n * Kr + k] : 0.0f;
    dst[i] = (__bf16)v;
}

// ---------------------------------------------------------------- LN + pad/roll/window-partition -> bf16 A
__global__ void ln_window_kernel(const float* __restrict__ X,
                                 const float* __restrict__ gw, const float* __restrict__ gb,
                                 __bf16* __restrict__ Aout,
                                 int C, int Kp, int D, int dp, int nwdim, int ws, int ss,
                                 int N, int Mreal, int Mpad) {
    int m = blockIdx.x * blockDim.x + threadIdx.x;
    if (m >= Mpad) return;
    __bf16* dst = Aout + (size_t)m * Kp;
    const float* srcp = nullptr;
    if (m < Mreal) {
        int win = m / N, t = m - win * N;
        int nw3 = nwdim * nwdim * nwdim;
        int b_ = win / nw3;
        int wrem = win - b_ * nw3;
        int wd = wrem / (nwdim * nwdim), wh = (wrem / nwdim) % nwdim, ww = wrem % nwdim;
        int td = t / (ws * ws), th = (t / ws) % ws, tw = t % ws;
        int zs = (wd * ws + td + ss) % dp;
        int ys = (wh * ws + th + ss) % dp;
        int xs = (ww * ws + tw + ss) % dp;
        if (zs < D && ys < D && xs < D)
            srcp = X + ((size_t)((b_ * D + zs) * D + ys) * D + xs) * C;
    }
    if (!srcp) {
        for (int k = 0; k < Kp; ++k) dst[k] = (__bf16)0.0f;
        return;
    }
    float mu = 0.f;
    for (int c = 0; c < C; ++c) mu += srcp[c];
    mu /= (float)C;
    float var = 0.f;
    for (int c = 0; c < C; ++c) { float d = srcp[c] - mu; var += d * d; }
    var /= (float)C;
    float inv = rsqrtf(var + 1e-5f);
    for (int k = 0; k < Kp; ++k)
        dst[k] = (k < C) ? (__bf16)((srcp[k] - mu) * inv * gw[k] + gb[k]) : (__bf16)0.0f;
}

// ---------------------------------------------------------------- LN (token-direct) -> bf16 A
__global__ void ln_pack_kernel(const float* __restrict__ X,
                               const float* __restrict__ gw, const float* __restrict__ gb,
                               __bf16* __restrict__ Aout, int C, int Kp, int M) {
    int m = blockIdx.x * blockDim.x + threadIdx.x;
    if (m >= M) return;
    const float* srcp = X + (size_t)m * C;
    __bf16* dst = Aout + (size_t)m * Kp;
    float mu = 0.f;
    for (int c = 0; c < C; ++c) mu += srcp[c];
    mu /= (float)C;
    float var = 0.f;
    for (int c = 0; c < C; ++c) { float d = srcp[c] - mu; var += d * d; }
    var /= (float)C;
    float inv = rsqrtf(var + 1e-5f);
    for (int k = 0; k < Kp; ++k)
        dst[k] = (k < C) ? (__bf16)((srcp[k] - mu) * inv * gw[k] + gb[k]) : (__bf16)0.0f;
}

// ---------------------------------------------------------------- patch-merge gather + LN(8C) -> bf16 A
__global__ void merge_ln_kernel(const float* __restrict__ X,
                                const float* __restrict__ gw, const float* __restrict__ gb,
                                __bf16* __restrict__ Aout, int C, int D, int Mm) {
    int m = blockIdx.x * blockDim.x + threadIdx.x;
    if (m >= Mm) return;
    int Dh = D >> 1;
    int xq = m % Dh, yq = (m / Dh) % Dh, zq = (m / (Dh * Dh)) % Dh, b_ = m / (Dh * Dh * Dh);
    int C8 = 8 * C;
    float mu = 0.f, sq = 0.f;
    for (int ii = 0; ii < 8; ++ii) {
        int i = ii >> 2, j = (ii >> 1) & 1, k = ii & 1;
        const float* s = X + ((size_t)((b_ * D + 2 * zq + i) * D + (2 * yq + j)) * D + (2 * xq + k)) * C;
        for (int c = 0; c < C; ++c) { float v = s[c]; mu += v; sq += v * v; }
    }
    mu /= (float)C8;
    float var = sq / (float)C8 - mu * mu;
    float inv = rsqrtf(var + 1e-5f);
    __bf16* dst = Aout + (size_t)m * C8;
    for (int ii = 0; ii < 8; ++ii) {
        int i = ii >> 2, j = (ii >> 1) & 1, k = ii & 1;
        const float* s = X + ((size_t)((b_ * D + 2 * zq + i) * D + (2 * yq + j)) * D + (2 * xq + k)) * C;
        for (int c = 0; c < C; ++c) {
            int col = ii * C + c;
            dst[col] = (__bf16)((s[c] - mu) * inv * gw[col] + gb[col]);
        }
    }
}

// ---------------------------------------------------------------- WMMA GEMM  C[M,N] = A[M,K] * W[N,K]^T
// MODE 0: store bf16 (+bias)            MODE 1: store bf16 (+bias, exact GELU)
// MODE 2: fp32 residual add (+bias)     MODE 3: fp32 residual add via window-reverse scatter (+bias)
// MODE 4: store fp32 direct (no bias)
template <int MODE>
__global__ __launch_bounds__(128) void gemm_bf16_kernel(
    const __bf16* __restrict__ A, const __bf16* __restrict__ W,
    const float* __restrict__ bias,
    __bf16* __restrict__ outB, float* __restrict__ outF,
    int Kp, int outStride, int Nreal, int Mpad, int Mreal,
    int Ntok, int wsz, int ssz, int dp, int D, int nwdim, int Cdim) {
    int wave = threadIdx.x >> 5;
    int lane = threadIdx.x & 31;
    int m0 = (blockIdx.x * 4 + wave) << 4;   // M tiles on grid.x (large)
    int n0 = blockIdx.y << 4;                // N tiles on grid.y (small)
    if (m0 >= Mpad) return;
    int half = lane >> 4;
    int r = lane & 15;
    const __bf16* arow = A + (size_t)(m0 + r) * Kp;
    const __bf16* brow = W + (size_t)(n0 + r) * Kp;
    v8f acc = {};
    for (int k0 = 0; k0 < Kp; k0 += 32) {
        int klo = k0 + half * 8;
        int khi = k0 + 16 + half * 8;
        v8bf alo = *(const v8bf*)(arow + klo);
        v8bf ahi = *(const v8bf*)(arow + khi);
        v8bf blo = *(const v8bf*)(brow + klo);
        v8bf bhi = *(const v8bf*)(brow + khi);
        v16bf av, bv;
#pragma unroll
        for (int i = 0; i < 8; ++i) {
            av[i] = alo[i]; av[8 + i] = ahi[i];
            bv[i] = blo[i]; bv[8 + i] = bhi[i];
        }
        acc = __builtin_amdgcn_wmma_f32_16x16x32_bf16(false, av, false, bv, (short)0, acc, false, false);
    }
    int col = n0 + r;
    float bval = 0.0f;
    if (bias && col < Nreal) bval = bias[col];
#pragma unroll
    for (int i = 0; i < 8; ++i) {
        int row = m0 + half * 8 + i;
        float v = acc[i] + bval;
        if (MODE == 0) {
            outB[(size_t)row * outStride + col] = (__bf16)v;
        } else if (MODE == 1) {
            outB[(size_t)row * outStride + col] = (__bf16)gelu_f(v);
        } else if (MODE == 2) {
            if (col < Nreal && row < Mreal)
                outF[(size_t)row * Cdim + col] += v;
        } else if (MODE == 3) {
            if (col < Nreal && row < Mreal) {
                int win = row / Ntok, t = row - win * Ntok;
                int nw3 = nwdim * nwdim * nwdim;
                int b_ = win / nw3;
                int wrem = win - b_ * nw3;
                int wd = wrem / (nwdim * nwdim), wh = (wrem / nwdim) % nwdim, ww = wrem % nwdim;
                int td = t / (wsz * wsz), th = (t / wsz) % wsz, tw = t % wsz;
                int zf = (wd * wsz + td + ssz) % dp;
                int yf = (wh * wsz + th + ssz) % dp;
                int xf = (ww * wsz + tw + ssz) % dp;
                if (zf < D && yf < D && xf < D) {
                    size_t dst = ((size_t)((b_ * D + zf) * D + yf) * D + xf) * Cdim + col;
                    outF[dst] += v;
                }
            }
        } else { // MODE 4
            if (row < Mreal)
                outF[(size_t)row * outStride + col] = v;
        }
    }
}

// ---------------------------------------------------------------- window attention (per window/head, flash softmax)
// head_dim == 8 always; rel-pos index & shifted-window mask computed analytically.
__global__ void attn_kernel(const __bf16* __restrict__ qkv, int qstride,
                            const float* __restrict__ rpb,
                            __bf16* __restrict__ outB, int ostride,
                            int C, int nh, int N, int nwdim, int ws, int ss, int dp,
                            int shifted) {
    extern __shared__ float smem[];
    int win = blockIdx.x / nh;
    int h = blockIdx.x - win * nh;
    float* qs = smem;
    float* ks = qs + N * 8;
    float* vs = ks + N * 8;
    int tid = threadIdx.x;
    for (int i = tid; i < N * 8; i += blockDim.x) {
        int t = i >> 3, e = i & 7;
        size_t row = (size_t)(win * N + t) * qstride;
        qs[i] = (float)qkv[row + h * 8 + e];
        ks[i] = (float)qkv[row + C + h * 8 + e];
        vs[i] = (float)qkv[row + 2 * C + h * 8 + e];
    }
    __syncthreads();
    int nw3 = nwdim * nwdim * nwdim;
    int wrem = win % nw3;
    int wd = wrem / (nwdim * nwdim), wh = (wrem / nwdim) % nwdim, ww = wrem % nwdim;
    const float scale = 0.3535533905932738f; // 8^-0.5
    for (int t = tid; t < N; t += blockDim.x) {
        float qv[8];
#pragma unroll
        for (int e = 0; e < 8; ++e) qv[e] = qs[t * 8 + e] * scale;
        int ad = t / 49, ah = (t / 7) % 7, aw = t % 7; // rel-pos coords (always 7-basis)
        int rq0 = 0, rq1 = 0, rq2 = 0;
        if (shifted) {
            int td = t / (ws * ws), th = (t / ws) % ws, tw = t % ws;
            rq0 = region_of(wd * ws + td, dp, ws, ss);
            rq1 = region_of(wh * ws + th, dp, ws, ss);
            rq2 = region_of(ww * ws + tw, dp, ws, ss);
        }
        float mx = -3.0e38f, sum = 0.f;
        float acc[8] = {0.f, 0.f, 0.f, 0.f, 0.f, 0.f, 0.f, 0.f};
        for (int m2 = 0; m2 < N; ++m2) {
            float s = 0.f;
#pragma unroll
            for (int e = 0; e < 8; ++e) s += qv[e] * ks[m2 * 8 + e];
            int bd = m2 / 49, bh = (m2 / 7) % 7, bw = m2 % 7;
            int idx = ((ad - bd + 6) * 13 + (ah - bh + 6)) * 13 + (aw - bw + 6);
            s += rpb[idx * nh + h];
            if (shifted) {
                int md = m2 / (ws * ws), mh = (m2 / ws) % ws, mw = m2 % ws;
                int rk0 = region_of(wd * ws + md, dp, ws, ss);
                int rk1 = region_of(wh * ws + mh, dp, ws, ss);
                int rk2 = region_of(ww * ws + mw, dp, ws, ss);
                if (rk0 != rq0 || rk1 != rq1 || rk2 != rq2) s -= 100.0f;
            }
            float mnew = fmaxf(mx, s);
            float corr = __expf(mx - mnew);
            float p = __expf(s - mnew);
            sum = sum * corr + p;
#pragma unroll
            for (int e = 0; e < 8; ++e) acc[e] = acc[e] * corr + p * vs[m2 * 8 + e];
            mx = mnew;
        }
        float invs = 1.0f / sum;
        size_t orow = (size_t)(win * N + t) * ostride;
#pragma unroll
        for (int e = 0; e < 8; ++e)
            outB[orow + h * 8 + e] = (__bf16)(acc[e] * invs);
    }
}

// ---------------------------------------------------------------- final: LN (no affine) + avg-pool(8) + L1 normalize
__global__ void final_head_kernel(const float* __restrict__ X, float* __restrict__ out) {
    __shared__ float red[512];
    int b_ = blockIdx.x, c = threadIdx.x; // blockDim = 384
    if (c < 128) red[384 + c] = 0.f;
    float acc = 0.f;
    for (int t = 0; t < 8; ++t) {
        float v = X[(size_t)(b_ * 8 + t) * 384 + c];
        __syncthreads();
        red[c] = v;
        __syncthreads();
        for (int s = 256; s > 0; s >>= 1) {
            if (c < s) red[c] += red[c + s];
            __syncthreads();
        }
        float mu = red[0] / 384.f;
        __syncthreads();
        float d = v - mu;
        red[c] = d * d;
        __syncthreads();
        for (int s = 256; s > 0; s >>= 1) {
            if (c < s) red[c] += red[c + s];
            __syncthreads();
        }
        float var = red[0] / 384.f;
        __syncthreads();
        acc += d * rsqrtf(var + 1e-5f);
    }
    acc *= 0.125f;
    __syncthreads();
    red[c] = fabsf(acc);
    __syncthreads();
    for (int s = 256; s > 0; s >>= 1) {
        if (c < s) red[c] += red[c + s];
        __syncthreads();
    }
    float denom = fmaxf(red[0], 1e-12f);
    out[b_ * 384 + c] = acc / denom;
}

// ================================================================ host
static inline int ceil_div(int a, int b) { return (a + b - 1) / b; }

extern "C" void kernel_launch(void* const* d_in, const int* in_sizes, int n_in,
                              void* d_out, int out_size, void* d_ws, size_t ws_size,
                              hipStream_t stream) {
    // ---- unpack inputs (setup_inputs insertion order, depth-first) ----
    int pi = 0;
    auto F = [&](void) { return (const float*)d_in[pi++]; };
    const float* x_in = F();
    const float* pe_w = F();
    const float* pe_b = F();
    const float* P[4][2][13]; // n1w,n1b,qkvw,qkvb,rpb,projw,projb,n2w,n2b,fc1w,fc1b,fc2w,fc2b
    const float* MG[4][3];    // nm_w, nm_b, red_w
    for (int li = 0; li < 4; ++li) {
        for (int bi = 0; bi < 2; ++bi)
            for (int j = 0; j < 13; ++j) P[li][bi][j] = F();
        for (int j = 0; j < 3; ++j) MG[li][j] = F();
    }

    // ---- layer constants ----
    static const int Cc[4]  = {24, 48, 96, 192};
    static const int NHc[4] = {3, 6, 12, 24};
    static const int Dc[4]  = {32, 16, 8, 4};
    static const int WSc[4] = {7, 7, 7, 4};
    static const int SSc[4] = {3, 3, 3, 0};     // shift of block1 (layer3 clamped to 0)
    static const int DPc[4] = {35, 21, 14, 4};
    static const int NWc[4] = {5, 3, 2, 1};
    static const int Ntc[4] = {343, 343, 343, 64};
    static const int Cpc[4] = {32, 64, 96, 192};     // K-pad of C (mult 32)
    static const int C3p[4] = {80, 144, 288, 576};   // N-pad of 3C (mult 16)
    static const int C4c[4] = {96, 192, 384, 768};
    static const int C8c[4] = {192, 384, 768, 1536};
    static const int C2c[4] = {48, 96, 192, 384};
    static const int Bsz = 2;

    // ---- workspace carve ----
    char* ws = (char*)d_ws;
    size_t off = 0;
    auto carve = [&](size_t bytes) {
        void* p = ws + off;
        off += (bytes + 255) & ~(size_t)255;
        return p;
    };
    float*  X0 = (float*)carve((size_t)65536 * 24 * 4);
    float*  X1 = (float*)carve((size_t)65536 * 24 * 4);
    __bf16* BA = (__bf16*)carve((size_t)85760 * 32 * 2);
    __bf16* BB = (__bf16*)carve((size_t)85760 * 80 * 2);
    __bf16* BC = (__bf16*)carve((size_t)85760 * 32 * 2);
    __bf16* BW = (__bf16*)carve((size_t)768 * 1536 * 2);

    // ---- patch embed ----
    {
        long total = (long)Bsz * 32 * 32 * 32 * 24;
        patch_embed_kernel<<<dim3((unsigned)((total + 255) / 256)), 256, 0, stream>>>(
            x_in, pe_w, pe_b, X0, Bsz, 64, 24);
    }

    float* X = X0;
    float* Xalt = X1;

    for (int li = 0; li < 4; ++li) {
        const int C = Cc[li], nh = NHc[li], D = Dc[li], wsz = WSc[li];
        const int dp = DPc[li], nw = NWc[li], N = Ntc[li];
        const int Cp = Cpc[li], c3p = C3p[li], C4 = C4c[li];
        const int B_ = Bsz * nw * nw * nw;
        const int Mreal = B_ * N;
        const int Mwp = ((Mreal + 15) / 16) * 16;
        const int Mtok = Bsz * D * D * D;

        for (int bi = 0; bi < 2; ++bi) {
            const float** p = P[li][bi];
            const int ss = (bi == 1) ? SSc[li] : 0;
            const int shifted = (ss > 0) ? 1 : 0;

            // LN1 + pad/roll/window-partition -> BA (Mwp x Cp) bf16
            ln_window_kernel<<<ceil_div(Mwp, 256), 256, 0, stream>>>(
                X, p[0], p[1], BA, C, Cp, D, dp, nw, wsz, ss, N, Mreal, Mwp);

            // qkv = BA @ qkv_w^T + b    (N = 3C)
            conv_weight_kernel<<<ceil_div(c3p * Cp, 256), 256, 0, stream>>>(
                p[2], BW, 3 * C, C, c3p, Cp);
            gemm_bf16_kernel<0><<<dim3(ceil_div(Mwp, 64), c3p / 16), 128, 0, stream>>>(
                BA, BW, p[3], BB, nullptr, Cp, c3p, 3 * C, Mwp, Mreal, 0, 0, 0, 0, 0, 0, 0);

            // attention -> BC (Mwp x Cp) bf16
            hipMemsetAsync(BC, 0, (size_t)Mwp * Cp * 2, stream);
            attn_kernel<<<B_ * nh, 256, (size_t)3 * N * 8 * 4, stream>>>(
                BB, c3p, p[4], BC, Cp, C, nh, N, nw, wsz, ss, dp, shifted);

            // proj + window-reverse + residual-add into X
            conv_weight_kernel<<<ceil_div(Cp * Cp, 256), 256, 0, stream>>>(
                p[5], BW, C, C, Cp, Cp);
            gemm_bf16_kernel<3><<<dim3(ceil_div(Mwp, 64), Cp / 16), 128, 0, stream>>>(
                BC, BW, p[6], nullptr, X, Cp, 0, C, Mwp, Mreal, N, wsz, ss, dp, D, nw, C);

            // MLP: LN2 -> BA, fc1(+GELU) -> BB, fc2 + residual -> X
            ln_pack_kernel<<<ceil_div(Mtok, 256), 256, 0, stream>>>(
                X, p[7], p[8], BA, C, Cp, Mtok);
            conv_weight_kernel<<<ceil_div(C4 * Cp, 256), 256, 0, stream>>>(
                p[9], BW, 4 * C, C, C4, Cp);
            gemm_bf16_kernel<1><<<dim3(ceil_div(Mtok, 64), C4 / 16), 128, 0, stream>>>(
                BA, BW, p[10], BB, nullptr, Cp, C4, 4 * C, Mtok, Mtok, 0, 0, 0, 0, 0, 0, 0);
            conv_weight_kernel<<<ceil_div(Cp * C4, 256), 256, 0, stream>>>(
                p[11], BW, C, 4 * C, Cp, C4);
            gemm_bf16_kernel<2><<<dim3(ceil_div(Mtok, 64), Cp / 16), 128, 0, stream>>>(
                BB, BW, p[12], nullptr, X, C4, 0, C, Mtok, Mtok, 0, 0, 0, 0, 0, 0, C);
        }

        // patch merging: gather(8 nbrs) + LN(8C) -> BA, then @ red_w^T -> Xalt
        const int Mm = Bsz * (D / 2) * (D / 2) * (D / 2);
        const int C8 = C8c[li], C2 = C2c[li];
        merge_ln_kernel<<<ceil_div(Mm, 256), 256, 0, stream>>>(
            X, MG[li][0], MG[li][1], BA, C, D, Mm);
        conv_weight_kernel<<<ceil_div(C2 * C8, 256), 256, 0, stream>>>(
            MG[li][2], BW, 2 * C, 8 * C, C2, C8);
        gemm_bf16_kernel<4><<<dim3(ceil_div(Mm, 64), C2 / 16), 128, 0, stream>>>(
            BA, BW, nullptr, nullptr, Xalt, C8, C2, 2 * C, Mm, Mm, 0, 0, 0, 0, 0, 0, 0);

        float* tmp = X; X = Xalt; Xalt = tmp;
    }

    // final head: X holds (2*8 tokens, 384)
    final_head_kernel<<<2, 384, 0, stream>>>(X, (float*)d_out);
}
